// TransformerDecoderOnly_53145925321474
// MI455X (gfx1250) — matrix operand order repaired and checked by reading, not compiled, not run on hardware
//
#include <hip/hip_runtime.h>
#include <hip/hip_bf16.h>

// ---------------------------------------------------------------------------
// GPT-2-style decoder forward for MI455X (gfx1250, wave32).
// GEMMs: v_wmma_f32_16x16x32_f16 (f16 operands, f32 accumulate).
// Tile staging: per-lane async global->LDS (ASYNCcnt) for A/Q/K/V tiles and
// TDM tensor_load_to_lds (TENSORcnt) for pre-transposed f16 weight tiles.
// ---------------------------------------------------------------------------

typedef __attribute__((ext_vector_type(16))) _Float16 v16h;
typedef __attribute__((ext_vector_type(8)))  float    v8f;
typedef unsigned int u32x4 __attribute__((ext_vector_type(4)));
typedef int          i32x4 __attribute__((ext_vector_type(4)));
typedef int          i32x8 __attribute__((ext_vector_type(8)));
typedef int          vi4   __attribute__((vector_size(16)));

#define PAD_ID 50256

#if defined(__gfx1250__) && __has_builtin(__builtin_amdgcn_global_load_async_to_lds_b128)
#define HAVE_ASYNC_LDS 1
#else
#define HAVE_ASYNC_LDS 0
#endif

#if defined(__gfx1250__) && __has_builtin(__builtin_amdgcn_tensor_load_to_lds)
#define HAVE_TDM 1
#else
#define HAVE_TDM 0
#endif

// 16-byte global -> LDS copy (async when available).
__device__ __forceinline__ void cp16_g2l(void* lds, const void* g) {
#if HAVE_ASYNC_LDS
  __attribute__((address_space(1))) void* gp =
      (__attribute__((address_space(1))) void*)(void*)g;
  __attribute__((address_space(3))) void* lp =
      (__attribute__((address_space(3))) void*)lds;
  __builtin_amdgcn_global_load_async_to_lds_b128(
      (__attribute__((address_space(1))) vi4*)gp,
      (__attribute__((address_space(3))) vi4*)lp, 0, 0);
#else
  *(uint4*)lds = *(const uint4*)g;
#endif
}

__device__ __forceinline__ void wait_async() {
#if HAVE_ASYNC_LDS
#if __has_builtin(__builtin_amdgcn_s_wait_asynccnt)
  __builtin_amdgcn_s_wait_asynccnt(0);
#else
  asm volatile("s_wait_asynccnt 0x0" ::: "memory");
#endif
#endif
}

__device__ __forceinline__ void wait_tensor() {
#if HAVE_TDM
#if __has_builtin(__builtin_amdgcn_s_wait_tensorcnt)
  __builtin_amdgcn_s_wait_tensorcnt(0);
#else
  asm volatile("s_wait_tensorcnt 0x0" ::: "memory");
#endif
#endif
}

__device__ __forceinline__ v8f wmma32(v16h a, v16h b, v8f c) {
  return __builtin_amdgcn_wmma_f32_16x16x32_f16(false, a, false, b, (short)0, c,
                                                false, false);
}

// A operand fragment (16x32 f16), tile row-major [row][k], stride in halves.
__device__ __forceinline__ v16h frag_a(const _Float16* base, int stride, int m0,
                                       int lane) {
  v16h a;
  const int m = m0 + (lane & 15);
  const int kh = (lane >> 4) * 8;
  const _Float16* p = base + m * stride;
#pragma unroll
  for (int j = 0; j < 8; ++j) {
    const int k = ((j & 4) ? 16 : 0) + kh + (j & 3) * 2;
    a[2 * j]     = p[k];
    a[2 * j + 1] = p[k + 1];
  }
  return a;
}

// B operand fragment (32x16 f16) from K-contiguous storage Bt[n][k].
__device__ __forceinline__ v16h frag_b(const _Float16* base, int stride, int n0,
                                       int lane) {
  v16h b;
  const int n = n0 + (lane & 15);
  const _Float16* p = base + n * stride + (lane >> 4) * 8;
#pragma unroll
  for (int j = 0; j < 8; ++j) {
    const int k = ((j & 4) ? 16 : 0) + (j & 3) * 2;
    b[2 * j]     = p[k];
    b[2 * j + 1] = p[k + 1];
  }
  return b;
}

// ---------------------------------------------------------------------------
// Generic batched GEMM: C[z] = act(A[z/aDiv] * Bt[z%bMod]^T + bias)
// A: f16 [M,K] row-major.  Bt: f16 [N,K] row-major (pre-transposed weights).
// ACT: 0 none, 1 exact GELU.  OUT: 0 f32 C[m*N+n], 1 f16 C[m*N+n],
// 2 f16 transposed C[n*ldc+m].
// ---------------------------------------------------------------------------
template <int BM, int BN, int WMn, int WNn, int ACT, int OUT>
__launch_bounds__(WMn * WNn * 32)
__global__ void gemm_kernel(const _Float16* __restrict__ A,
                            const _Float16* __restrict__ Bt,
                            void* __restrict__ Cv,
                            const float* __restrict__ bias,
                            int M, int N, int K, int ldc,
                            long aStride, int aDiv,
                            long bStride, int bMod,
                            long cStride) {
  constexpr int BK = 32;
  constexpr int NT = WMn * WNn * 32;
  constexpr int LA = BK + 8;  // padded LDS row stride (halves): 80B = 5x16B

  __shared__ _Float16 As[BM][LA];
  __shared__ _Float16 Bs[BN][LA];

  const int z = blockIdx.z;
  A  += (long)(z / aDiv) * aStride;
  Bt += (long)(z % bMod) * bStride;
  float*    Cf = (float*)Cv + (long)z * cStride;
  _Float16* Ch = (_Float16*)Cv + (long)z * cStride;

  const int tid  = threadIdx.x;
  const int mBlk = blockIdx.y * BM;
  const int nBlk = blockIdx.x * BN;
  const int wave = tid >> 5, lane = tid & 31;
  const int wm = (wave / WNn) * 32;
  const int wn = (wave % WNn) * 32;
  const int half = lane >> 4, ln = lane & 15;

  const _Float16* BtBlk = Bt + (long)nBlk * K;
  const int rowsB = (N - nBlk < BN) ? (N - nBlk) : BN;

#if HAVE_TDM
  // Invariant part of the TDM 2D descriptor for the B tile:
  // tile BK x rowsB halves out of tensor [N,K] f16, padded LDS rows
  // (pad_interval code 3 = every 16 DWORDs, pad_amount code 3 = 4 DWORDs).
  i32x8 g1;
  g1[0] = (int)((1u << 16) | (1u << 20) | (3u << 22) | (3u << 25));
  g1[1] = (int)(((unsigned)K & 0xffffu) << 16);
  g1[2] = (int)((((unsigned)K >> 16) & 0xffffu) |
                (((unsigned)rowsB & 0xffffu) << 16));
  g1[3] = (int)((((unsigned)rowsB >> 16) & 0xffffu) | ((unsigned)BK << 16));
  g1[4] = (int)((unsigned)rowsB & 0xffffu);
  g1[5] = (int)(unsigned)K;  // tensor_dim0_stride (elements)
  g1[6] = 0;
  g1[7] = 0;
  const i32x4 zg4 = {0, 0, 0, 0};
  const i32x8 zg8 = {0, 0, 0, 0, 0, 0, 0, 0};
  const unsigned ldsB = (unsigned)(unsigned long long)(size_t)&Bs[0][0];
#endif

  v8f acc[2][2];
  const v8f vz = {};
#pragma unroll
  for (int i = 0; i < 2; ++i)
#pragma unroll
    for (int j = 0; j < 2; ++j) acc[i][j] = vz;

  for (int k0 = 0; k0 < K; k0 += BK) {
    // ---- stage A tile: BM x BK halves, 16B chunks per lane (async) ----
    for (int c = tid; c < BM * (BK / 8); c += NT) {
      const int r = c >> 2, cc = (c & 3) * 8;
      const int gm = mBlk + r;
      if (gm < M) cp16_g2l(&As[r][cc], A + (long)gm * K + k0 + cc);
    }
    // ---- stage B tile: rowsB x BK halves ----
#if HAVE_TDM
    {
      const unsigned long long ga =
          (unsigned long long)(size_t)(BtBlk + k0);
      u32x4 g0;
      g0[0] = 1u;  // count=1
      g0[1] = ldsB;
      g0[2] = (unsigned)ga;
      g0[3] = (unsigned)((ga >> 32) & 0x01FFFFFFull) | 0x80000000u;  // type=2
      if (wave == 0)
        __builtin_amdgcn_tensor_load_to_lds(g0, g1, zg4, zg4, zg8, 0);
    }
#else
    for (int c = tid; c < BN * (BK / 8); c += NT) {
      const int r = c >> 2, cc = (c & 3) * 8;
      if (r < rowsB) cp16_g2l(&Bs[r][cc], BtBlk + (long)r * K + k0 + cc);
    }
#endif
    if (wave == 0) wait_tensor();
    wait_async();
    __syncthreads();

    v16h af[2], bf[2];
    af[0] = frag_a(&As[0][0], LA, wm, lane);
    af[1] = frag_a(&As[0][0], LA, wm + 16, lane);
    bf[0] = frag_b(&Bs[0][0], LA, wn, lane);
    bf[1] = frag_b(&Bs[0][0], LA, wn + 16, lane);
#pragma unroll
    for (int i = 0; i < 2; ++i)
#pragma unroll
      for (int j = 0; j < 2; ++j) acc[i][j] = wmma32(af[i], bf[j], acc[i][j]);
    __syncthreads();
  }

  // ---- epilogue ----
#pragma unroll
  for (int i = 0; i < 2; ++i) {
#pragma unroll
    for (int j = 0; j < 2; ++j) {
#pragma unroll
      for (int r = 0; r < 8; ++r) {
        const int gm = mBlk + wm + i * 16 + r + 8 * half;
        const int gn = nBlk + wn + j * 16 + ln;
        if (gm < M && gn < N) {
          float vv = acc[i][j][r];
          if (bias) vv += bias[gn];
          if (ACT == 1)
            vv = 0.5f * vv * (1.0f + erff(vv * 0.70710678118654752440f));
          if (OUT == 0)
            Cf[(long)gm * N + gn] = vv;
          else if (OUT == 1)
            Ch[(long)gm * N + gn] = (_Float16)vv;
          else
            Ch[(long)gn * ldc + gm] = (_Float16)vv;
        }
      }
    }
  }
}

// ---------------------------------------------------------------------------
// Flash attention per (b,h): 64-query tile/block, streaming causal key tiles.
// q16,k16: f16 [B*H, T, DK]; vT16: f16 [B*H, DK, T]; out ctx16: f16 [B*T, D].
// ---------------------------------------------------------------------------
__launch_bounds__(128)
__global__ void attn_kernel(const _Float16* __restrict__ Q,
                            const _Float16* __restrict__ Kg,
                            const _Float16* __restrict__ Vt16,
                            const int* __restrict__ tok,
                            _Float16* __restrict__ ctx,
                            int Hh, int T, int DK, int D) {
  constexpr int LA = 64 + 8;  // 144B rows (multiple of 16B)
  __shared__ _Float16 Qs[64][LA];
  __shared__ _Float16 Ks[64][LA];   // [key][d]  (B operand for S)
  __shared__ _Float16 Vs[64][LA];   // [d][key]  (B operand for O)
  __shared__ _Float16 Ps[64][LA];   // probabilities (A operand for O)
  __shared__ float    Ss[64][65];
  __shared__ float rowm[64], rowl[64], alpha_s[64];

  const int z = blockIdx.y;
  const int b = z / Hh, h = z % Hh;
  const _Float16* Qp = Q    + (long)z * T * DK;
  const _Float16* Kp = Kg   + (long)z * T * DK;
  const _Float16* Vp = Vt16 + (long)z * DK * T;
  const int qt = blockIdx.x;
  const int q0 = qt * 64;

  const int tid = threadIdx.x;
  const int wave = tid >> 5, lane = tid & 31;
  const int wm = (wave >> 1) * 32, wn = (wave & 1) * 32;
  const int half = lane >> 4, ln = lane & 15;
  const float scale = rsqrtf((float)DK);

  // stage Q tile once (64 rows x 64 halves = 8 chunks/row)
  for (int c = tid; c < 64 * 8; c += 128) {
    const int r = c >> 3, cc = (c & 7) * 8;
    cp16_g2l(&Qs[r][cc], Qp + (long)(q0 + r) * DK + cc);
  }
  if (tid < 64) { rowm[tid] = -1e30f; rowl[tid] = 0.f; }

  v8f o[2][2];
  const v8f vz = {};
#pragma unroll
  for (int i = 0; i < 2; ++i)
#pragma unroll
    for (int j = 0; j < 2; ++j) o[i][j] = vz;

  for (int kt = 0; kt <= qt; ++kt) {
    const int k0 = kt * 64;
    for (int c = tid; c < 64 * 8; c += 128) {
      const int r = c >> 3, cc = (c & 7) * 8;
      cp16_g2l(&Ks[r][cc], Kp + (long)(k0 + r) * DK + cc);
      cp16_g2l(&Vs[r][cc], Vp + (long)r * T + k0 + cc);
    }
    wait_async();
    __syncthreads();

    // S = Q * K^T
    v8f s[2][2];
#pragma unroll
    for (int i = 0; i < 2; ++i)
#pragma unroll
      for (int j = 0; j < 2; ++j) s[i][j] = vz;
    for (int kk = 0; kk < 64; kk += 32) {
      v16h af[2], bf[2];
      af[0] = frag_a(&Qs[0][kk], LA, wm, lane);
      af[1] = frag_a(&Qs[0][kk], LA, wm + 16, lane);
      bf[0] = frag_b(&Ks[0][kk], LA, wn, lane);
      bf[1] = frag_b(&Ks[0][kk], LA, wn + 16, lane);
#pragma unroll
      for (int i = 0; i < 2; ++i)
#pragma unroll
        for (int j = 0; j < 2; ++j) s[i][j] = wmma32(af[i], bf[j], s[i][j]);
    }
#pragma unroll
    for (int i = 0; i < 2; ++i)
#pragma unroll
      for (int j = 0; j < 2; ++j)
#pragma unroll
        for (int r = 0; r < 8; ++r) {
          const int qm = wm + i * 16 + r + 8 * half;
          const int kn = wn + j * 16 + ln;
          const int gq = q0 + qm, gk = k0 + kn;
          const bool vis = (gk <= gq) && (tok[b * T + gk] != PAD_ID);
          Ss[qm][kn] = vis ? s[i][j][r] * scale : -1e30f;
        }
    __syncthreads();

    if (tid < 64) {
      const int r = tid;
      float mt = -1e30f;
      for (int c = 0; c < 64; ++c) mt = fmaxf(mt, Ss[r][c]);
      const float mold = rowm[r];
      const float mnew = fmaxf(mold, mt);
      const float al = __expf(mold - mnew);
      float psum = 0.f;
      for (int c = 0; c < 64; ++c) {
        const float p = __expf(Ss[r][c] - mnew);
        Ps[r][c] = (_Float16)p;
        psum += p;
      }
      rowl[r] = rowl[r] * al + psum;
      rowm[r] = mnew;
      alpha_s[r] = al;
    }
    __syncthreads();

    float alv[2][8];
#pragma unroll
    for (int i = 0; i < 2; ++i)
#pragma unroll
      for (int r = 0; r < 8; ++r)
        alv[i][r] = alpha_s[wm + i * 16 + r + 8 * half];
#pragma unroll
    for (int i = 0; i < 2; ++i)
#pragma unroll
      for (int j = 0; j < 2; ++j)
#pragma unroll
        for (int r = 0; r < 8; ++r) o[i][j][r] *= alv[i][r];

    for (int kk = 0; kk < 64; kk += 32) {
      v16h af[2], bf[2];
      af[0] = frag_a(&Ps[0][kk], LA, wm, lane);
      af[1] = frag_a(&Ps[0][kk], LA, wm + 16, lane);
      bf[0] = frag_b(&Vs[0][kk], LA, wn, lane);
      bf[1] = frag_b(&Vs[0][kk], LA, wn + 16, lane);
#pragma unroll
      for (int i = 0; i < 2; ++i)
#pragma unroll
        for (int j = 0; j < 2; ++j) o[i][j] = wmma32(af[i], bf[j], o[i][j]);
    }
    __syncthreads();
  }

#pragma unroll
  for (int i = 0; i < 2; ++i)
#pragma unroll
    for (int j = 0; j < 2; ++j)
#pragma unroll
      for (int r = 0; r < 8; ++r) {
        const int qm = wm + i * 16 + r + 8 * half;
        const int dn = wn + j * 16 + ln;
        ctx[((long)(b * T + q0 + qm)) * D + h * DK + dn] =
            (_Float16)(o[i][j][r] / rowl[qm]);
      }
}

// ---------------------------------------------------------------------------
// Tiled transpose + f32->f16 convert: out[z][n*K+k] = (f16) in[z][k*N+n]
// ---------------------------------------------------------------------------
__launch_bounds__(256)
__global__ void transp_cvt_kernel(const float* __restrict__ in,
                                  _Float16* __restrict__ out,
                                  int K, int N, long inStride, long outStride) {
  __shared__ float tile[32][33];
  const float* ip = in + (long)blockIdx.z * inStride;
  _Float16* op = out + (long)blockIdx.z * outStride;
  const int kb = blockIdx.y * 32, nb = blockIdx.x * 32;
  const int tx = threadIdx.x & 31, ty = threadIdx.x >> 5;
#pragma unroll
  for (int i = 0; i < 32; i += 8) {
    const int kk = kb + ty + i, nn = nb + tx;
    tile[ty + i][tx] = (kk < K && nn < N) ? ip[(long)kk * N + nn] : 0.f;
  }
  __syncthreads();
#pragma unroll
  for (int i = 0; i < 32; i += 8) {
    const int nn = nb + ty + i, kk = kb + tx;
    if (nn < N && kk < K) op[(long)nn * K + kk] = (_Float16)tile[tx][ty + i];
  }
}

// ---------------------------------------------------------------------------
// Embedding: x = tok_emb[id]*sqrt(D) + pos_emb[t]; writes f32 + f16 copies.
// ---------------------------------------------------------------------------
__global__ void embed_kernel(const int* __restrict__ tok,
                             const float* __restrict__ te,
                             const float* __restrict__ pe,
                             float* __restrict__ x, _Float16* __restrict__ x16,
                             int T, int D) {
  const int row = blockIdx.x;
  const int t = row % T;
  const int id = tok[row];
  const float sc = sqrtf((float)D);
  for (int d = threadIdx.x; d < D; d += blockDim.x) {
    const float v = (id == PAD_ID) ? 0.f : te[(long)id * D + d];
    const float r = v * sc + pe[(long)t * D + d];
    x[(long)row * D + d] = r;
    x16[(long)row * D + d] = (_Float16)r;
  }
}

// ---------------------------------------------------------------------------
// Fused residual + LayerNorm: x = LN(x + y)*g + b ; also writes f16 copy.
// ---------------------------------------------------------------------------
__launch_bounds__(256)
__global__ void add_ln_kernel(float* __restrict__ x, _Float16* __restrict__ x16,
                              const float* __restrict__ y,
                              const float* __restrict__ g,
                              const float* __restrict__ bta, int D) {
  __shared__ float buf[256];
  const int row = blockIdx.x;
  float* xr = x + (long)row * D;
  _Float16* hr = x16 + (long)row * D;
  const float* yr = y + (long)row * D;

  float v[2];
  float s = 0.f;
#pragma unroll
  for (int i = 0; i < 2; ++i) {
    const int d = threadIdx.x + i * 256;
    v[i] = xr[d] + yr[d];
    s += v[i];
  }
  buf[threadIdx.x] = s;
  __syncthreads();
  for (int off = 128; off > 0; off >>= 1) {
    if (threadIdx.x < off) buf[threadIdx.x] += buf[threadIdx.x + off];
    __syncthreads();
  }
  const float mu = buf[0] / (float)D;
  __syncthreads();

  float s2 = 0.f;
#pragma unroll
  for (int i = 0; i < 2; ++i) {
    const float dlt = v[i] - mu;
    s2 += dlt * dlt;
  }
  buf[threadIdx.x] = s2;
  __syncthreads();
  for (int off = 128; off > 0; off >>= 1) {
    if (threadIdx.x < off) buf[threadIdx.x] += buf[threadIdx.x + off];
    __syncthreads();
  }
  const float rstd = rsqrtf(buf[0] / (float)D + 1e-5f);
#pragma unroll
  for (int i = 0; i < 2; ++i) {
    const int d = threadIdx.x + i * 256;
    const float r = (v[i] - mu) * rstd * g[d] + bta[d];
    xr[d] = r;
    hr[d] = (_Float16)r;
  }
}

// ---------------------------------------------------------------------------
// Host-side orchestration
// ---------------------------------------------------------------------------
extern "C" void kernel_launch(void* const* d_in, const int* in_sizes, int n_in,
                              void* d_out, int out_size, void* d_ws,
                              size_t ws_size, hipStream_t stream) {
  (void)in_sizes; (void)n_in; (void)out_size; (void)ws_size;

  const int*   tok  = (const int*)d_in[0];
  const float* te   = (const float*)d_in[1];
  const float* pe   = (const float*)d_in[2];
  const float* WQ   = (const float*)d_in[3];
  const float* WK   = (const float*)d_in[4];
  const float* WV   = (const float*)d_in[5];
  const float* WO   = (const float*)d_in[6];
  const float* ln1g = (const float*)d_in[7];
  const float* ln1b = (const float*)d_in[8];
  const float* fc1w = (const float*)d_in[9];
  const float* fc1b = (const float*)d_in[10];
  const float* fc2w = (const float*)d_in[11];
  const float* fc2b = (const float*)d_in[12];
  const float* ln2g = (const float*)d_in[13];
  const float* ln2b = (const float*)d_in[14];
  const float* lmh  = (const float*)d_in[15];
  float* out = (float*)d_out;

  constexpr int Bc = 2, Tc = 1024, Dc = 512, Hc = 8, DKc = 64, DFFc = 2048,
                Lc = 6, Vc = 50257;
  const int M = Bc * Tc;  // 2048

  // workspace allocator (256B aligned slices)
  size_t off = 0;
  auto alloc = [&](size_t bytes) -> void* {
    void* p = (char*)d_ws + off;
    off += (bytes + 255) & ~(size_t)255;
    return p;
  };
  float* x        = (float*)alloc((size_t)M * Dc * 4);
  float* attnout  = (float*)alloc((size_t)M * Dc * 4);
  float* ffnout   = (float*)alloc((size_t)M * Dc * 4);
  _Float16* x16   = (_Float16*)alloc((size_t)M * Dc * 2);
  _Float16* ctx16 = (_Float16*)alloc((size_t)M * Dc * 2);
  _Float16* q16   = (_Float16*)alloc((size_t)M * Dc * 2);
  _Float16* k16   = (_Float16*)alloc((size_t)M * Dc * 2);
  _Float16* vT16  = (_Float16*)alloc((size_t)M * Dc * 2);
  _Float16* h16   = (_Float16*)alloc((size_t)M * DFFc * 2);
  _Float16* wq16t = (_Float16*)alloc((size_t)Lc * Hc * Dc * DKc * 2);
  _Float16* wk16t = (_Float16*)alloc((size_t)Lc * Hc * Dc * DKc * 2);
  _Float16* wv16t = (_Float16*)alloc((size_t)Lc * Hc * Dc * DKc * 2);
  _Float16* wo16t = (_Float16*)alloc((size_t)Lc * Dc * Dc * 2);
  _Float16* fc1t  = (_Float16*)alloc((size_t)Lc * Dc * DFFc * 2);
  _Float16* fc2t  = (_Float16*)alloc((size_t)Lc * DFFc * Dc * 2);
  _Float16* lmt   = (_Float16*)alloc((size_t)Vc * Dc * 2);

  // ---- one-time weight transpose+convert to f16 [N,K] ----
  transp_cvt_kernel<<<dim3(DKc / 32, Dc / 32, Lc * Hc), 256, 0, stream>>>(
      WQ, wq16t, Dc, DKc, (long)Dc * DKc, (long)DKc * Dc);
  transp_cvt_kernel<<<dim3(DKc / 32, Dc / 32, Lc * Hc), 256, 0, stream>>>(
      WK, wk16t, Dc, DKc, (long)Dc * DKc, (long)DKc * Dc);
  transp_cvt_kernel<<<dim3(DKc / 32, Dc / 32, Lc * Hc), 256, 0, stream>>>(
      WV, wv16t, Dc, DKc, (long)Dc * DKc, (long)DKc * Dc);
  transp_cvt_kernel<<<dim3(Dc / 32, Dc / 32, Lc), 256, 0, stream>>>(
      WO, wo16t, Dc, Dc, (long)Dc * Dc, (long)Dc * Dc);
  transp_cvt_kernel<<<dim3(DFFc / 32, Dc / 32, Lc), 256, 0, stream>>>(
      fc1w, fc1t, Dc, DFFc, (long)Dc * DFFc, (long)DFFc * Dc);
  transp_cvt_kernel<<<dim3(Dc / 32, DFFc / 32, Lc), 256, 0, stream>>>(
      fc2w, fc2t, DFFc, Dc, (long)DFFc * Dc, (long)Dc * DFFc);
  transp_cvt_kernel<<<dim3((Vc + 31) / 32, Dc / 32, 1), 256, 0, stream>>>(
      lmh, lmt, Dc, Vc, 0, 0);

  embed_kernel<<<M, 256, 0, stream>>>(tok, te, pe, x, x16, Tc, Dc);

  for (int l = 0; l < Lc; ++l) {
    // QKV projections, batched over (b,h): [T,D] x [D,DK] per head
    const dim3 gq(1, Tc / 64, Bc * Hc);
    gemm_kernel<64, 64, 2, 2, 0, 1><<<gq, 128, 0, stream>>>(
        x16, wq16t + (size_t)l * Hc * DKc * Dc, q16, nullptr,
        Tc, DKc, Dc, DKc, (long)Tc * Dc, Hc, (long)DKc * Dc, Hc,
        (long)Tc * DKc);
    gemm_kernel<64, 64, 2, 2, 0, 1><<<gq, 128, 0, stream>>>(
        x16, wk16t + (size_t)l * Hc * DKc * Dc, k16, nullptr,
        Tc, DKc, Dc, DKc, (long)Tc * Dc, Hc, (long)DKc * Dc, Hc,
        (long)Tc * DKc);
    // V written transposed: vT16[z][d][t]
    gemm_kernel<64, 64, 2, 2, 0, 2><<<gq, 128, 0, stream>>>(
        x16, wv16t + (size_t)l * Hc * DKc * Dc, vT16, nullptr,
        Tc, DKc, Dc, Tc, (long)Tc * Dc, Hc, (long)DKc * Dc, Hc,
        (long)DKc * Tc);

    // Flash attention -> ctx16 (heads concatenated)
    const dim3 ga(Tc / 64, Bc * Hc);
    attn_kernel<<<ga, 128, 0, stream>>>(q16, k16, vT16, tok, ctx16, Hc, Tc,
                                        DKc, Dc);

    // Output projection (f32 out for residual)
    const dim3 go(Dc / 128, M / 64, 1);
    gemm_kernel<64, 128, 2, 4, 0, 0><<<go, 256, 0, stream>>>(
        ctx16, wo16t + (size_t)l * Dc * Dc, attnout, nullptr,
        M, Dc, Dc, Dc, 0, 1, 0, 1, 0);

    add_ln_kernel<<<M, 256, 0, stream>>>(x, x16, attnout,
                                         ln1g + (size_t)l * Dc,
                                         ln1b + (size_t)l * Dc, Dc);

    // FFN
    const dim3 g1(DFFc / 128, M / 64, 1);
    gemm_kernel<64, 128, 2, 4, 1, 1><<<g1, 256, 0, stream>>>(
        x16, fc1t + (size_t)l * DFFc * Dc, h16, fc1b + (size_t)l * DFFc,
        M, DFFc, Dc, DFFc, 0, 1, 0, 1, 0);

    const dim3 g2(Dc / 128, M / 64, 1);
    gemm_kernel<64, 128, 2, 4, 0, 0><<<g2, 256, 0, stream>>>(
        h16, fc2t + (size_t)l * Dc * DFFc, ffnout, fc2b + (size_t)l * Dc,
        M, Dc, DFFc, Dc, 0, 1, 0, 1, 0);

    add_ln_kernel<<<M, 256, 0, stream>>>(x, x16, ffnout,
                                         ln2g + (size_t)l * Dc,
                                         ln2b + (size_t)l * Dc, Dc);
  }

  // LM head: [2048,512] x [512,50257] -> f32 logits
  const dim3 gl((Vc + 127) / 128, M / 64, 1);
  gemm_kernel<64, 128, 2, 4, 0, 0><<<gl, 256, 0, stream>>>(
      x16, lmt, out, nullptr, M, Vc, Dc, Vc, 0, 1, 0, 1, 0);
}
